// MappingNet_89824946028570
// MI455X (gfx1250) — matrix operand order repaired and checked by reading, not compiled, not run on hardware
//
#include <hip/hip_runtime.h>
#include <math.h>

// ---------------------------------------------------------------------------
// MappingNet on MI455X (gfx1250): routed expert MLP, bf16 WMMA GEMMs,
// double-buffered LDS, async global->LDS staging for bf16 activations.
//   L1: act1 = gelu(x @ W1 + b1)            [4096,512]x[512,1024]
//   L2: s    = gelu(act1 @ W2 + b2)         [4096,1024]x[1024,1024]
//   LA: h    = gelu(gather(s) @ Wa[g] + ba) grouped per-genre GEMM
//   LB: out  =       gather(h) @ Wb[g] + bb grouped per-genre GEMM -> d_out
// ---------------------------------------------------------------------------

constexpr int BATCH = 4096;
constexpr int IN_D  = 512;
constexpr int D     = 1024;
constexpr int NG    = 16;

constexpr int STRIDE = 72;           // LDS row stride in bf16 elems (144 B, 16B-aligned)
constexpr int BUFSZ  = 128 * STRIDE; // one tile buffer (bf16 elems)

typedef __attribute__((ext_vector_type(16))) __bf16 v16bf;
typedef __attribute__((ext_vector_type(8)))  float  v8f;

union Frag { v16bf v; unsigned int u[8]; };

static __device__ __forceinline__ unsigned short f2bf(float f) {
    unsigned int u = __float_as_uint(f);
    u += 0x7FFFu + ((u >> 16) & 1u);           // round-to-nearest-even
    return (unsigned short)(u >> 16);
}

static __device__ __forceinline__ float gelu_exact(float x) {
    return 0.5f * x * (1.0f + erff(x * 0.70710678118654752f));
}

static __device__ __forceinline__ void wait_async0() {
#if __has_builtin(__builtin_amdgcn_s_wait_asynccnt)
    __builtin_amdgcn_s_wait_asynccnt(0);
#else
    asm volatile("s_wait_asynccnt 0x0" ::: "memory");
#endif
}

// -------------------- routing: bucket rows by genre ------------------------
__global__ void zero_cnt_kernel(int* cnt) {
    if (threadIdx.x < NG) cnt[threadIdx.x] = 0;
}

__global__ void build_perm_kernel(const int* __restrict__ genre,
                                  int* __restrict__ cnt,
                                  int* __restrict__ perm) {
    int b = blockIdx.x * blockDim.x + threadIdx.x;
    if (b < BATCH) {
        int g = genre[b];
        int slot = atomicAdd(&cnt[g], 1);
        perm[g * BATCH + slot] = b;
    }
}

// -------------------- fused GEMM + bias (+GELU) ----------------------------
// Block tile 128x128, BK=64, 256 threads = 8 waves (2x4), wave tile 64x32.
// Double-buffered LDS, one barrier per BK=64 iteration, 16 WMMAs/iteration.
// A path: f32 (layer 1, VGPR convert) or bf16 (async global->LDS copy).
// W path: f32 -> float4 loads -> bf16 convert -> K-transposed LDS stores,
// committed after the first 8 WMMAs so global latency hides under the XDL.
template <int K, bool A_F32, bool GATHER, bool GELU_BF16_OUT>
__global__ __launch_bounds__(256) void gemm_wmma_kernel(
    const void*  __restrict__ Aptr,
    const float* __restrict__ Wall,
    const float* __restrict__ ball,
    void*        __restrict__ Outptr,
    const int*   __restrict__ perm_all,
    const int*   __restrict__ cnt_all)
{
    __shared__ unsigned short ldsA[2 * BUFSZ];   // [buf][row][k]
    __shared__ unsigned short ldsB[2 * BUFSZ];   // [buf][n][k] (K-transposed W)

    const int t = threadIdx.x;
    const int g = GATHER ? (int)blockIdx.z : 0;
    const float* W    = Wall + (size_t)g * D * D;
    const float* bias = ball + (size_t)g * D;
    const int count   = GATHER ? cnt_all[g] : BATCH;
    const int* permg  = GATHER ? (perm_all + g * BATCH) : nullptr;

    const int m0 = blockIdx.y * 128;
    if (m0 >= count) return;                     // uniform exit: empty tile
    const int n0 = blockIdx.x * 128;

    const int lane = t & 31, lm = lane & 15, lh = lane >> 4;
    const int wave = t >> 5, wm = wave >> 2, wn = wave & 3;

    // Routed global row per A-staging slot.
    //  bf16/async: 4 slots of 8 cols -> row = t/8 + 32r, col = (t&7)*8
    //  f32       : 8 slots of 4 cols -> row = t/16 + 16r, col = (t&15)*4
    int rowsA[8];
#pragma unroll
    for (int r = 0; r < 8; ++r) {
        int i = A_F32 ? (m0 + (t >> 4) + 16 * r) : (m0 + (t >> 3) + 32 * (r & 3));
        if (GATHER) {
            int ic = (i < count) ? i : (count - 1);
            rowsA[r] = permg[ic];
        } else {
            rowsA[r] = i;
        }
    }

    // ---- staging helpers -------------------------------------------------
    auto issueB = [&](int k0, float4* breg) {
#pragma unroll
        for (int r = 0; r < 8; ++r) {
            int s = t + 256 * r;
            int kr = s >> 5, nc4 = s & 31;
            breg[r] = *(const float4*)&W[(size_t)(k0 + kr) * D + n0 + nc4 * 4];
        }
    };
    auto commitB = [&](unsigned short* bB, const float4* breg) {
#pragma unroll
        for (int r = 0; r < 8; ++r) {
            int s = t + 256 * r;
            int kr = s >> 5, nc = (s & 31) * 4;
            bB[(nc + 0) * STRIDE + kr] = f2bf(breg[r].x);
            bB[(nc + 1) * STRIDE + kr] = f2bf(breg[r].y);
            bB[(nc + 2) * STRIDE + kr] = f2bf(breg[r].z);
            bB[(nc + 3) * STRIDE + kr] = f2bf(breg[r].w);
        }
    };
    auto issueA_f32 = [&](int k0, float4* areg) {
#pragma unroll
        for (int r = 0; r < 8; ++r) {
            int col = (t & 15) * 4;
            areg[r] = *(const float4*)&((const float*)Aptr)[(size_t)rowsA[r] * K + k0 + col];
        }
    };
    auto commitA_f32 = [&](unsigned short* bA, const float4* areg) {
#pragma unroll
        for (int r = 0; r < 8; ++r) {
            int row = (t >> 4) + 16 * r;
            int col = (t & 15) * 4;
            uint2 p;
            p.x = (unsigned)f2bf(areg[r].x) | ((unsigned)f2bf(areg[r].y) << 16);
            p.y = (unsigned)f2bf(areg[r].z) | ((unsigned)f2bf(areg[r].w) << 16);
            *(uint2*)&bA[row * STRIDE + col] = p;
        }
    };
    auto issueA_async = [&](int k0, unsigned short* bA) {
#pragma unroll
        for (int r = 0; r < 4; ++r) {
            int row = (t >> 3) + 32 * r;
            int col = (t & 7) * 8;
            const unsigned short* gp =
                (const unsigned short*)Aptr + (size_t)rowsA[r] * K + k0 + col;
            unsigned ldsa = (unsigned)(unsigned long long)&bA[row * STRIDE + col];
            asm volatile("global_load_async_to_lds_b128 %0, %1, off"
                         :: "v"(ldsa), "v"((unsigned long long)gp)
                         : "memory");
        }
    };

    // ---- fragment load / WMMA helpers ------------------------------------
    auto loadFrags = [&](const unsigned short* bA, const unsigned short* bB,
                         int ks, Frag* af, Frag* bfr) {
#pragma unroll
        for (int mi = 0; mi < 4; ++mi)
#pragma unroll
            for (int v = 0; v < 8; ++v) {
                int kk = 32 * ks + ((v < 4) ? (lh * 8 + 2 * v)
                                            : (16 + lh * 8 + 2 * (v - 4)));
                af[mi].u[v] = *(const unsigned int*)
                    &bA[(wm * 64 + mi * 16 + lm) * STRIDE + kk];
            }
#pragma unroll
        for (int ni = 0; ni < 2; ++ni)
#pragma unroll
            for (int v = 0; v < 8; ++v) {
                int kk = 32 * ks + ((v < 4) ? (lh * 8 + 2 * v)
                                            : (16 + lh * 8 + 2 * (v - 4)));
                bfr[ni].u[v] = *(const unsigned int*)
                    &bB[(wn * 32 + ni * 16 + lm) * STRIDE + kk];
            }
    };

    v8f acc[4][2] = {};
    auto mma8 = [&](const Frag* af, const Frag* bfr) {
#pragma unroll
        for (int mi = 0; mi < 4; ++mi)
#pragma unroll
            for (int ni = 0; ni < 2; ++ni)
                acc[mi][ni] = __builtin_amdgcn_wmma_f32_16x16x32_bf16(
                    false, af[mi].v, false, bfr[ni].v,
                    (short)0, acc[mi][ni], false, false);
    };

    // ---- prologue: stage tile 0 into buffer 0 ----------------------------
    constexpr int NK = K / 64;
    {
        float4 breg[8], areg[8];
        issueB(0, breg);
        if (A_F32) issueA_f32(0, areg);
        else       issueA_async(0, ldsA);
        commitB(ldsB, breg);
        if (A_F32) commitA_f32(ldsA, areg);
    }

    // ---- main pipelined loop ---------------------------------------------
    for (int it = 0; it < NK; ++it) {
        if (!A_F32) wait_async0();               // publish async LDS writes
        __syncthreads();

        const int cur = it & 1;
        unsigned short* bAc = ldsA + cur * BUFSZ;
        unsigned short* bBc = ldsB + cur * BUFSZ;
        unsigned short* bAn = ldsA + (cur ^ 1) * BUFSZ;
        unsigned short* bBn = ldsB + (cur ^ 1) * BUFSZ;
        const bool hasNext = (it + 1 < NK);
        const int k0n = (it + 1) * 64;

        Frag af[4], bfr[2];
        float4 breg[8], areg[8];

        loadFrags(bAc, bBc, 0, af, bfr);
        if (hasNext) {                           // issue next tile's traffic
            issueB(k0n, breg);
            if (A_F32) issueA_f32(k0n, areg);
            else       issueA_async(k0n, bAn);
        }
        mma8(af, bfr);                           // k-step 0 (overlaps loads)

        loadFrags(bAc, bBc, 1, af, bfr);
        if (hasNext) {                           // commit after XDL is busy
            commitB(bBn, breg);
            if (A_F32) commitA_f32(bAn, areg);
        }
        mma8(af, bfr);                           // k-step 1
    }

    // ---- epilogue: bias (+GELU), scatter to original rows ----------------
#pragma unroll
    for (int ni = 0; ni < 2; ++ni) {
        int nG = n0 + wn * 32 + ni * 16 + lm;
        float bv = bias[nG];
#pragma unroll
        for (int mi = 0; mi < 4; ++mi) {
#pragma unroll
            for (int r = 0; r < 8; ++r) {
                int i = m0 + wm * 64 + mi * 16 + r + 8 * lh;  // C/D layout
                if (!GATHER || i < count) {
                    int orow = GATHER ? permg[i] : i;
                    float val = acc[mi][ni][r] + bv;
                    if (GELU_BF16_OUT) {
                        val = gelu_exact(val);
                        ((unsigned short*)Outptr)[(size_t)orow * D + nG] = f2bf(val);
                    } else {
                        ((float*)Outptr)[(size_t)orow * D + nG] = val;
                    }
                }
            }
        }
    }
}

// ---------------------------------------------------------------------------
extern "C" void kernel_launch(void* const* d_in, const int* in_sizes, int n_in,
                              void* d_out, int out_size, void* d_ws, size_t ws_size,
                              hipStream_t stream) {
    (void)in_sizes; (void)n_in; (void)out_size; (void)ws_size;

    const float* x     = (const float*)d_in[0];
    const int*   genre = (const int*)  d_in[1];
    const float* W1    = (const float*)d_in[2];
    const float* b1    = (const float*)d_in[3];
    const float* W2    = (const float*)d_in[4];
    const float* b2    = (const float*)d_in[5];
    const float* Wa    = (const float*)d_in[6];
    const float* ba    = (const float*)d_in[7];
    const float* Wb    = (const float*)d_in[8];
    const float* bb    = (const float*)d_in[9];

    char* ws = (char*)d_ws;
    int* cnt  = (int*)ws;                          // 16 ints
    int* perm = (int*)(ws + 256);                  // 16 * 4096 ints = 256 KB
    unsigned short* act1 = (unsigned short*)(ws + (1 << 20));   // 8 MB bf16
    unsigned short* sbf  = act1 + (size_t)BATCH * D;            // 8 MB bf16
    unsigned short* hbf  = sbf  + (size_t)BATCH * D;            // 8 MB bf16

    // routing
    zero_cnt_kernel<<<1, 32, 0, stream>>>(cnt);
    build_perm_kernel<<<BATCH / 256, 256, 0, stream>>>(genre, cnt, perm);

    // trunk
    gemm_wmma_kernel<IN_D, true,  false, true ><<<dim3(8, 32, 1),  256, 0, stream>>>(
        x,    W1, b1, act1, nullptr, nullptr);
    gemm_wmma_kernel<D,    false, false, true ><<<dim3(8, 32, 1),  256, 0, stream>>>(
        act1, W2, b2, sbf,  nullptr, nullptr);

    // routed experts (grid.z = genre)
    gemm_wmma_kernel<D,    false, true,  true ><<<dim3(8, 32, NG), 256, 0, stream>>>(
        sbf, Wa, ba, hbf,   perm, cnt);
    gemm_wmma_kernel<D,    false, true,  false><<<dim3(8, 32, NG), 256, 0, stream>>>(
        hbf, Wb, bb, d_out, perm, cnt);
}
